// LTM_87368224735393
// MI455X (gfx1250) — matrix operand order repaired and checked by reading, not compiled, not run on hardware
//
#include <hip/hip_runtime.h>
#include <hip/hip_bf16.h>

// ---------------------------------------------------------------------------
// TNP transformer forward (loss scalar) for MI455X / gfx1250.
// GEMMs + attention on v_wmma_f32_16x16x32_bf16 (wave32 WMMA), with
// double-buffered LDS staging via global_load_async_to_lds_b128 (ASYNCcnt).
// ---------------------------------------------------------------------------

#define DEV __device__ __forceinline__

typedef __attribute__((ext_vector_type(16))) __bf16 v16bf;
typedef __attribute__((ext_vector_type(8)))  float  v8f;

#define BATCH 4
#define NCTX 384
#define NTGT 384
#define SEQ  (NCTX + 2*NTGT)   // 1152
#define DM   512
#define FFD  2048
#define NH   8
#define HD   64
#define NL   4
#define MTOK (BATCH*SEQ)       // 4608
#define MZ   (BATCH*NTGT)      // 1536
#define INPAD 96               // in_dim 65 padded to multiple of 32
#define NEGV (-1.0e9f)

DEV unsigned short f2b(float f) {           // fp32 -> bf16 (RNE)
  unsigned u = __float_as_uint(f);
  u = (u + 0x7fffu + ((u >> 16) & 1u)) >> 16;
  return (unsigned short)u;
}
DEV float b2f(unsigned short h) { return __uint_as_float(((unsigned)h) << 16); }
DEV float geluf(float x) { return 0.5f * x * (1.0f + erff(x * 0.70710678118654752f)); }

union AFrag { v16bf v; uint4 q[2]; unsigned short u[16]; };

// --- CDNA5 async global->LDS copy (16B per lane), tracked by ASYNCcnt -------
DEV void async_b128(const unsigned short* gptr, unsigned short* lptr) {
  unsigned loff = (unsigned)(size_t)lptr;   // low 32 bits of generic addr = LDS offset
  asm volatile("global_load_async_to_lds_b128 %0, %1, off"
               :: "v"(loff), "v"(gptr) : "memory");
}
DEV void wait_async3() { asm volatile("s_wait_asynccnt 0x3" ::: "memory"); }
DEV void wait_async2() { asm volatile("s_wait_asynccnt 0x2" ::: "memory"); }
DEV void wait_async0() { asm volatile("s_wait_asynccnt 0x0" ::: "memory"); }

// ---------------------------------------------------------------------------
// Weight convert: fp32 W[K][N] -> bf16 Wt[N][Kp]  (transposed, K zero-padded)
// ---------------------------------------------------------------------------
__global__ void conv_wt(const float* __restrict__ W, unsigned short* __restrict__ out,
                        int K, int N, int Kp) {
  int idx = blockIdx.x * blockDim.x + threadIdx.x;
  if (idx >= N * Kp) return;
  int n = idx / Kp, k = idx % Kp;
  out[idx] = f2b(k < K ? W[(size_t)k * N + n] : 0.0f);
}

// ---------------------------------------------------------------------------
// Build padded input tokens (B,S,96) bf16
// ---------------------------------------------------------------------------
__global__ void build_input(const float* __restrict__ xc, const float* __restrict__ yc,
                            const float* __restrict__ xt, const float* __restrict__ yt,
                            const float* __restrict__ rc, const float* __restrict__ rt,
                            const float* __restrict__ maskt, unsigned short* __restrict__ out) {
  int idx = blockIdx.x * blockDim.x + threadIdx.x;
  if (idx >= MTOK * INPAD) return;
  int row = idx / INPAD, c = idx % INPAD;
  int b = row / SEQ, s = row % SEQ;
  float v = 0.0f;
  if (s < NCTX) {
    int base = (b * NCTX + s) * 32;
    if (c < 32)       v = xc[base + c] * rc[base + c];
    else if (c < 64)  v = rc[base + c - 32];
    else if (c == 64) v = yc[b * NCTX + s];
  } else if (s < NCTX + NTGT) {
    int t = s - NCTX; int base = (b * NTGT + t) * 32;
    if (c < 32)       v = xt[base + c] * rt[base + c];
    else if (c < 64)  v = rt[base + c - 32];
    else if (c == 64) v = yt[b * NTGT + t];
  } else {
    int t = s - NCTX - NTGT; int base = (b * NTGT + t) * 32;
    if (c < 32)       v = xt[base + c] * rt[base + c] * maskt[base + c];
    else if (c < 64)  v = rt[base + c - 32] * maskt[base + c - 32];
  }
  out[idx] = f2b(v);
}

// ---------------------------------------------------------------------------
// GEMM: C[M,N] = A[M,K](bf16) x Wt[N,K](bf16)^T + bias, optional GELU.
// 256 threads (8 waves), 128x64 tile, K-step 32, double-buffered LDS with
// async global->LDS copies overlapping WMMA compute.
// ---------------------------------------------------------------------------
#define GF_GELU 1
#define GF_F32  2
#define GF_ACC  4
#define GF_BF16 8

__global__ __launch_bounds__(256)
void gemm_bf16(const unsigned short* __restrict__ A,
               const unsigned short* __restrict__ Wt,
               const float* __restrict__ bias,
               float* __restrict__ outF, unsigned short* __restrict__ outH,
               int M, int N, int K, int flags) {
  __shared__ alignas(16) unsigned short sA[2][128 * 32];
  __shared__ alignas(16) unsigned short sB[2][64 * 32];
  const int tid = threadIdx.x;
  const int w = tid >> 5, lane = tid & 31;
  const int hf = lane >> 4, ln = lane & 15;
  const int m0 = blockIdx.y * 128, n0 = blockIdx.x * 64;
  const int rA = tid >> 1, hA = tid & 1;     // A staging: 128 rows x 2 halves (16 elems)
  const int rB = tid >> 2, pB = tid & 3;     // B staging: 64 rows x 4 parts (8 elems)

  v8f acc[4];
  for (int i = 0; i < 4; i++)
    for (int j = 0; j < 8; j++) acc[i][j] = 0.0f;

  auto stage = [&](int kk, int bufi) {       // 3 async b128 per thread
    const unsigned short* ga = A + (size_t)(m0 + rA) * K + kk + 16 * hA;
    unsigned short* la = &sA[bufi][rA * 32 + 16 * hA];
    async_b128(ga, la);
    async_b128(ga + 8, la + 8);
    const unsigned short* gb = Wt + (size_t)(n0 + rB) * K + kk + 8 * pB;
    async_b128(gb, &sB[bufi][rB * 32 + 8 * pB]);
  };

  stage(0, 0);
  int buf = 0;
  for (int kk = 0; kk < K; kk += 32) {
    const bool more = (kk + 32 < K);
    if (more) { stage(kk + 32, buf ^ 1); wait_async3(); }
    else      { wait_async0(); }
    __syncthreads();
    // A fragment (16x32): row m contiguous run at k=16*hf
    AFrag a;
    a.q[0] = *(const uint4*)(&sA[buf][(16 * w + ln) * 32 + 16 * hf]);
    a.q[1] = *(const uint4*)(&sA[buf][(16 * w + ln) * 32 + 16 * hf + 8]);
    for (int ns = 0; ns < 4; ns++) {
      AFrag b;  // B fragment (32x16): column n contiguous run at k=16*hf
      const uint4* bp = (const uint4*)(&sB[buf][(16 * ns + ln) * 32 + 16 * hf]);
      b.q[0] = bp[0]; b.q[1] = bp[1];
      acc[ns] = __builtin_amdgcn_wmma_f32_16x16x32_bf16(
          false, a.v, false, b.v, (short)0, acc[ns], false, false);
    }
    __syncthreads();
    buf ^= 1;
  }
  // epilogue: C layout element (r = i + 8*hf, c = ln)
  for (int ns = 0; ns < 4; ns++) {
    int col = n0 + 16 * ns + ln;
    float bv = bias ? bias[col] : 0.0f;
    for (int i = 0; i < 8; i++) {
      int row = m0 + 16 * w + i + 8 * hf;
      float v = acc[ns][i] + bv;
      if (flags & GF_GELU) v = geluf(v);
      size_t idx = (size_t)row * N + col;
      if (flags & GF_ACC)      outF[idx] += v;
      else if (flags & GF_F32) outF[idx] = v;
      if (flags & GF_BF16)     outH[idx] = f2b(v);
    }
  }
}

// ---------------------------------------------------------------------------
// LayerNorm (+optional residual):  y = LN(base + resid) * g + b
// ---------------------------------------------------------------------------
__global__ __launch_bounds__(256)
void ln_kernel(const float* __restrict__ base, const float* __restrict__ resid,
               const float* __restrict__ g, const float* __restrict__ bb,
               float* __restrict__ xf, unsigned short* __restrict__ xh, int Mrows) {
  int w = threadIdx.x >> 5, lane = threadIdx.x & 31;
  int row = blockIdx.x * 8 + w;
  if (row >= Mrows) return;
  const float* bp = base + (size_t)row * DM;
  const float* rp = resid ? resid + (size_t)row * DM : nullptr;
  float vals[16], s = 0.0f, s2 = 0.0f;
  for (int j = 0; j < 16; j++) {
    int c = lane + 32 * j;
    float v = bp[c] + (rp ? rp[c] : 0.0f);
    vals[j] = v; s += v; s2 += v * v;
  }
  for (int off = 16; off >= 1; off >>= 1) {
    s  += __shfl_xor(s,  off, 32);
    s2 += __shfl_xor(s2, off, 32);
  }
  float mean = s / (float)DM;
  float var  = s2 / (float)DM - mean * mean;
  float rs = rsqrtf(var + 1e-5f);
  for (int j = 0; j < 16; j++) {
    int c = lane + 32 * j;
    float y = (vals[j] - mean) * rs * g[c] + bb[c];
    xf[(size_t)row * DM + c] = y;
    xh[(size_t)row * DM + c] = f2b(y);
  }
}

// ---------------------------------------------------------------------------
// Split qkv (M x 1536 bf16) into Q/K/V (B,H,S,64) bf16. Q pre-scaled 1/sqrt(64).
// ---------------------------------------------------------------------------
__global__ void split_qkv(const unsigned short* __restrict__ qkv,
                          unsigned short* __restrict__ Q,
                          unsigned short* __restrict__ Kd,
                          unsigned short* __restrict__ V) {
  int idx = blockIdx.x * blockDim.x + threadIdx.x;
  if (idx >= MTOK * DM) return;
  int row = idx / DM, c = idx % DM;
  int h = c / HD, d = c % HD;
  int b = row / SEQ, s = row % SEQ;
  size_t src = (size_t)row * (3 * DM);
  size_t dst = ((size_t)(b * NH + h) * SEQ + s) * HD + d;
  Q[dst]  = f2b(b2f(qkv[src + c]) * 0.125f);
  Kd[dst] = qkv[src + DM + c];
  V[dst]  = qkv[src + 2 * DM + c];
}

// structural attention mask (computed analytically)
DEV float maskval(int q, int k) {
  if (k < NCTX) return 0.0f;
  if (k == q)   return 0.0f;                 // diagonal override
  if (q < NCTX) return NEGV;
  if (q < NCTX + NTGT) {
    if (k < NCTX + NTGT) return (k <= q) ? 0.0f : NEGV;   // causal incl diag
    return NEGV;
  }
  if (k < NCTX + NTGT) return ((k - NCTX) < (q - NCTX - NTGT)) ? 0.0f : NEGV;
  return NEGV;
}

// ---------------------------------------------------------------------------
// Flash attention: 4 waves = 64 q rows; double-buffered K/V staging
// (K via async-to-LDS, V via register prefetch + transposed LDS store);
// scores via WMMA with mask as C operand; online softmax; P@V via WMMA.
// ---------------------------------------------------------------------------
__global__ __launch_bounds__(128)
void flash_attn(const unsigned short* __restrict__ Q,
                const unsigned short* __restrict__ Kg,
                const unsigned short* __restrict__ Vg,
                unsigned short* __restrict__ att) {
  __shared__ alignas(16) unsigned short sK[2][32 * 64];   // [s][d]
  __shared__ alignas(16) unsigned short sV[2][64 * 32];   // [d][s] (transposed)
  __shared__ alignas(16) unsigned short sP[4][16 * 32];
  int tid = threadIdx.x, w = tid >> 5, lane = tid & 31;
  int hf = lane >> 4, ln = lane & 15;
  int bh = blockIdx.x, b = bh / NH, h = bh % NH;
  int q0 = blockIdx.y * 64 + 16 * w;
  const unsigned short* Qb = Q  + (size_t)bh * SEQ * HD;
  const unsigned short* Kb = Kg + (size_t)bh * SEQ * HD;
  const unsigned short* Vb = Vg + (size_t)bh * SEQ * HD;

  AFrag aq[2];  // Q fragments for d-chunks 0..31 / 32..63 (loaded once)
  {
    const unsigned short* qr = Qb + (size_t)(q0 + ln) * HD;
    aq[0].q[0] = *(const uint4*)(qr + 16 * hf);
    aq[0].q[1] = *(const uint4*)(qr + 16 * hf + 8);
    aq[1].q[0] = *(const uint4*)(qr + 32 + 16 * hf);
    aq[1].q[1] = *(const uint4*)(qr + 40 + 16 * hf);
  }
  v8f oacc[4];
  for (int i = 0; i < 4; i++)
    for (int j = 0; j < 8; j++) oacc[i][j] = 0.0f;
  float mrow[8], lsum[8];
  for (int i = 0; i < 8; i++) { mrow[i] = -3.0e38f; lsum[i] = 0.0f; }

  const int sRow = tid >> 2, sPart = tid & 3;   // staging: 32 rows x 4 chunks of 16
  auto issueK = [&](int kt, int bufi) {          // 2 async b128 per thread
    const unsigned short* gk = Kb + (size_t)(kt + sRow) * HD + sPart * 16;
    unsigned short* lk = &sK[bufi][sRow * 64 + sPart * 16];
    async_b128(gk, lk);
    async_b128(gk + 8, lk + 8);
  };
  uint4 v0, v1;
  auto loadV = [&](int kt) {                     // register prefetch of V rows
    const uint4* vs = (const uint4*)(Vb + (size_t)(kt + sRow) * HD + sPart * 16);
    v0 = vs[0]; v1 = vs[1];
  };

  issueK(0, 0);
  loadV(0);
  int buf = 0;
  for (int kt = 0; kt < SEQ; kt += 32) {
    // transpose-store prefetched V rows into sV[buf]
    {
      union { uint4 q; unsigned short u[8]; } t0, t1;
      t0.q = v0; t1.q = v1;
      for (int j = 0; j < 8; j++) {
        sV[buf][(sPart * 16 + j) * 32 + sRow]     = t0.u[j];
        sV[buf][(sPart * 16 + 8 + j) * 32 + sRow] = t1.u[j];
      }
    }
    const bool more = (kt + 32 < SEQ);
    if (more) { issueK(kt + 32, buf ^ 1); loadV(kt + 32); wait_async2(); }
    else      { wait_async0(); }
    __syncthreads();
    // scores for two 16-wide s subtiles, mask injected as the C operand
    v8f sc[2];
    for (int st = 0; st < 2; st++) {
      v8f c;
      for (int i = 0; i < 8; i++)
        c[i] = maskval(q0 + i + 8 * hf, kt + st * 16 + ln);
      AFrag bk;
      const uint4* kp = (const uint4*)(&sK[buf][(st * 16 + ln) * 64 + 16 * hf]);
      bk.q[0] = kp[0]; bk.q[1] = kp[1];
      c = __builtin_amdgcn_wmma_f32_16x16x32_bf16(false, aq[0].v, false, bk.v, (short)0, c, false, false);
      const uint4* kp2 = (const uint4*)(&sK[buf][(st * 16 + ln) * 64 + 32 + 16 * hf]);
      bk.q[0] = kp2[0]; bk.q[1] = kp2[1];
      c = __builtin_amdgcn_wmma_f32_16x16x32_bf16(false, aq[1].v, false, bk.v, (short)0, c, false, false);
      sc[st] = c;
    }
    // online softmax over the 32 new columns
    for (int i = 0; i < 8; i++) {
      float rmax = fmaxf(sc[0][i], sc[1][i]);
      rmax = fmaxf(rmax, __shfl_xor(rmax, 1, 32));
      rmax = fmaxf(rmax, __shfl_xor(rmax, 2, 32));
      rmax = fmaxf(rmax, __shfl_xor(rmax, 4, 32));
      rmax = fmaxf(rmax, __shfl_xor(rmax, 8, 32));
      float mnew  = fmaxf(mrow[i], rmax);
      float alpha = __expf(mrow[i] - mnew);
      float p0 = __expf(sc[0][i] - mnew);
      float p1 = __expf(sc[1][i] - mnew);
      float psum = p0 + p1;
      psum += __shfl_xor(psum, 1, 32);
      psum += __shfl_xor(psum, 2, 32);
      psum += __shfl_xor(psum, 4, 32);
      psum += __shfl_xor(psum, 8, 32);
      lsum[i] = lsum[i] * alpha + psum;
      mrow[i] = mnew;
      for (int nd = 0; nd < 4; nd++) oacc[nd][i] *= alpha;
      int r = i + 8 * hf;                      // C-layout -> LDS (A-layout bounce)
      sP[w][r * 32 + ln]      = f2b(p0);
      sP[w][r * 32 + 16 + ln] = f2b(p1);
    }
    // P @ V  (K = 32 s values)
    AFrag ap;
    ap.q[0] = *(const uint4*)(&sP[w][ln * 32 + 16 * hf]);
    ap.q[1] = *(const uint4*)(&sP[w][ln * 32 + 16 * hf + 8]);
    for (int nd = 0; nd < 4; nd++) {
      AFrag bv;
      const uint4* vp = (const uint4*)(&sV[buf][(nd * 16 + ln) * 32 + 16 * hf]);
      bv.q[0] = vp[0]; bv.q[1] = vp[1];
      oacc[nd] = __builtin_amdgcn_wmma_f32_16x16x32_bf16(
          false, ap.v, false, bv.v, (short)0, oacc[nd], false, false);
    }
    __syncthreads();
    buf ^= 1;
  }
  for (int nd = 0; nd < 4; nd++)
    for (int i = 0; i < 8; i++) {
      int q = q0 + i + 8 * hf;
      float ov = oacc[nd][i] / lsum[i];
      att[(size_t)(b * SEQ + q) * DM + h * HD + nd * 16 + ln] = f2b(ov);
    }
}

// ---------------------------------------------------------------------------
// Gather last NT tokens per batch
// ---------------------------------------------------------------------------
__global__ void gather_z(const unsigned short* __restrict__ xh, unsigned short* __restrict__ zh) {
  int idx = blockIdx.x * blockDim.x + threadIdx.x;
  if (idx >= MZ * DM) return;
  int row = idx / DM, c = idx % DM;
  int b = row / NTGT, t = row % NTGT;
  zh[idx] = xh[(size_t)(b * SEQ + NCTX + NTGT + t) * DM + c];
}

// ---------------------------------------------------------------------------
// Final 2048 -> (mean, logstd) projection + per-token Gaussian NLL term
// ---------------------------------------------------------------------------
__global__ __launch_bounds__(256)
void pred_loss(const unsigned short* __restrict__ p2, const float* __restrict__ Wo,
               const float* __restrict__ bo, const float* __restrict__ yt,
               float* __restrict__ lp) {
  __shared__ float r0[256], r1[256];
  int t = blockIdx.x;
  const unsigned short* row = p2 + (size_t)t * FFD;
  float a0 = 0.0f, a1 = 0.0f;
  for (int c = threadIdx.x; c < FFD; c += 256) {
    float v = b2f(row[c]);
    a0 += v * Wo[c * 2 + 0];
    a1 += v * Wo[c * 2 + 1];
  }
  r0[threadIdx.x] = a0; r1[threadIdx.x] = a1;
  __syncthreads();
  for (int s = 128; s > 0; s >>= 1) {
    if (threadIdx.x < s) { r0[threadIdx.x] += r0[threadIdx.x + s]; r1[threadIdx.x] += r1[threadIdx.x + s]; }
    __syncthreads();
  }
  if (threadIdx.x == 0) {
    float mean = r0[0] + bo[0];
    float logstd = r1[0] + bo[1];
    float d = (yt[t] - mean) / __expf(logstd);
    lp[t] = -0.5f * d * d - logstd - 0.918938533204672741f;  // 0.5*log(2*pi)
  }
}

__global__ __launch_bounds__(256)
void reduce_loss(const float* __restrict__ lp, float* __restrict__ out) {
  __shared__ float r[256];
  float s = 0.0f;
  for (int i = threadIdx.x; i < MZ; i += 256) s += lp[i];
  r[threadIdx.x] = s;
  __syncthreads();
  for (int st = 128; st > 0; st >>= 1) {
    if (threadIdx.x < st) r[threadIdx.x] += r[threadIdx.x + st];
    __syncthreads();
  }
  if (threadIdx.x == 0) out[0] = -r[0] / (float)MZ;
}

// ---------------------------------------------------------------------------
extern "C" void kernel_launch(void* const* d_in, const int* in_sizes, int n_in,
                              void* d_out, int out_size, void* d_ws, size_t ws_size,
                              hipStream_t stream) {
  (void)in_sizes; (void)n_in; (void)out_size; (void)ws_size;
  // ---- inputs (setup_inputs dict order; params in insertion order) ----
  const float* xc    = (const float*)d_in[0];
  const float* yc    = (const float*)d_in[1];
  const float* xt    = (const float*)d_in[2];
  const float* yt    = (const float*)d_in[3];
  const float* rc    = (const float*)d_in[4];
  const float* rt    = (const float*)d_in[5];
  const float* maskt = (const float*)d_in[6];
  const float* eW0 = (const float*)d_in[7];  const float* eb0 = (const float*)d_in[8];
  const float* eW1 = (const float*)d_in[9];  const float* eb1 = (const float*)d_in[10];
  const float* eWo = (const float*)d_in[11]; const float* ebo = (const float*)d_in[12];
  const float* eWp = (const float*)d_in[13]; const float* ebp = (const float*)d_in[14];
  const float* eg  = (const float*)d_in[15]; const float* ebe = (const float*)d_in[16];
  const float *Wqkv[NL], *bqkv[NL], *Wol[NL], *bol[NL], *g1[NL], *bb1[NL],
              *g2[NL], *bb2[NL], *Wf1[NL], *bf1[NL], *Wf2[NL], *bf2[NL];
  for (int l = 0; l < NL; l++) {
    int bse = 17 + 12 * l;
    Wqkv[l] = (const float*)d_in[bse + 0]; bqkv[l] = (const float*)d_in[bse + 1];
    Wol[l]  = (const float*)d_in[bse + 2]; bol[l]  = (const float*)d_in[bse + 3];
    g1[l]   = (const float*)d_in[bse + 4]; bb1[l]  = (const float*)d_in[bse + 5];
    g2[l]   = (const float*)d_in[bse + 6]; bb2[l]  = (const float*)d_in[bse + 7];
    Wf1[l]  = (const float*)d_in[bse + 8]; bf1[l]  = (const float*)d_in[bse + 9];
    Wf2[l]  = (const float*)d_in[bse + 10]; bf2[l] = (const float*)d_in[bse + 11];
  }
  const float* pW0 = (const float*)d_in[65]; const float* pb0 = (const float*)d_in[66];
  const float* pW1 = (const float*)d_in[67]; const float* pb1 = (const float*)d_in[68];
  const float* pWo = (const float*)d_in[69]; const float* pbo = (const float*)d_in[70];

  // ---- workspace layout (deterministic bump allocator) ----
  char* wsb = (char*)d_ws; size_t off = 0;
  auto alloc = [&](size_t bytes) -> char* {
    char* p = wsb + off; off += (bytes + 255) & ~(size_t)255; return p;
  };
  unsigned short* inp96 = (unsigned short*)alloc((size_t)MTOK * INPAD * 2);
  float*          xf    = (float*)         alloc((size_t)MTOK * DM * 4);
  unsigned short* xh    = (unsigned short*)alloc((size_t)MTOK * DM * 2);
  float*          tmpf  = (float*)         alloc((size_t)MTOK * DM * 4);
  unsigned short* big1  = (unsigned short*)alloc((size_t)MTOK * FFD * 2);
  unsigned short* big2  = (unsigned short*)alloc((size_t)MTOK * FFD * 2);
  // QKV + att alias big2 (big2 dead after embedding)
  const size_t HSZ = (size_t)BATCH * NH * SEQ * HD;
  unsigned short* Qb  = big2;
  unsigned short* Kb  = big2 + HSZ;
  unsigned short* Vb  = big2 + 2 * HSZ;
  unsigned short* att = big2 + 3 * HSZ;
  // pred buffers alias big1 (dead after last FF)
  unsigned short* zh = big1;
  unsigned short* p1 = big1 + (size_t)MZ * DM;
  unsigned short* p2 = p1 + (size_t)MZ * FFD;
  float* lp = tmpf;  // tmpf dead after last LN
  // bf16 transposed weights
  unsigned short* W0t = (unsigned short*)alloc((size_t)2048 * 96 * 2);
  unsigned short* W1t = (unsigned short*)alloc((size_t)2048 * 2048 * 2);
  unsigned short* Wot = (unsigned short*)alloc((size_t)512 * 2048 * 2);
  unsigned short* Wpt = (unsigned short*)alloc((size_t)512 * 96 * 2);
  unsigned short *Wqkvt[NL], *Wolt[NL], *Wf1t[NL], *Wf2t[NL];
  for (int l = 0; l < NL; l++) {
    Wqkvt[l] = (unsigned short*)alloc((size_t)1536 * 512 * 2);
    Wolt[l]  = (unsigned short*)alloc((size_t)512 * 512 * 2);
    Wf1t[l]  = (unsigned short*)alloc((size_t)2048 * 512 * 2);
    Wf2t[l]  = (unsigned short*)alloc((size_t)512 * 2048 * 2);
  }
  unsigned short* Wp0t = (unsigned short*)alloc((size_t)2048 * 512 * 2);
  unsigned short* Wp1t = (unsigned short*)alloc((size_t)2048 * 2048 * 2);

  auto convW = [&](const float* W, unsigned short* dst, int K, int N, int Kp) {
    int tot = N * Kp;
    conv_wt<<<dim3((tot + 255) / 256), dim3(256), 0, stream>>>(W, dst, K, N, Kp);
  };
  auto gemm = [&](const unsigned short* A, const unsigned short* Wt, const float* bias,
                  float* oF, unsigned short* oH, int M, int N, int K, int flags) {
    gemm_bf16<<<dim3(N / 64, M / 128), dim3(256), 0, stream>>>(A, Wt, bias, oF, oH, M, N, K, flags);
  };

  // ---- weight conversion ----
  convW(eW0, W0t, 65, 2048, 96);
  convW(eW1, W1t, 2048, 2048, 2048);
  convW(eWo, Wot, 2048, 512, 2048);
  convW(eWp, Wpt, 65, 512, 96);
  for (int l = 0; l < NL; l++) {
    convW(Wqkv[l], Wqkvt[l], 512, 1536, 512);
    convW(Wol[l],  Wolt[l],  512, 512, 512);
    convW(Wf1[l],  Wf1t[l],  512, 2048, 512);
    convW(Wf2[l],  Wf2t[l],  2048, 512, 2048);
  }
  convW(pW0, Wp0t, 512, 2048, 512);
  convW(pW1, Wp1t, 2048, 2048, 2048);

  // ---- build input tokens ----
  {
    int tot = MTOK * INPAD;
    build_input<<<dim3((tot + 255) / 256), dim3(256), 0, stream>>>(
        xc, yc, xt, yt, rc, rt, maskt, inp96);
  }

  // ---- embedding residual MLP ----
  gemm(inp96, W0t, eb0, nullptr, big1, MTOK, 2048, 96,   GF_GELU | GF_BF16);
  gemm(big1,  W1t, eb1, nullptr, big2, MTOK, 2048, 2048, GF_GELU | GF_BF16);
  gemm(big2,  Wot, ebo, tmpf, nullptr, MTOK, 512, 2048,  GF_F32);
  gemm(inp96, Wpt, ebp, tmpf, nullptr, MTOK, 512, 96,    GF_ACC);
  ln_kernel<<<dim3(MTOK / 8), dim3(256), 0, stream>>>(tmpf, nullptr, eg, ebe, xf, xh, MTOK);

  // ---- encoder layers ----
  for (int l = 0; l < NL; l++) {
    gemm(xh, Wqkvt[l], bqkv[l], nullptr, big1, MTOK, 1536, 512, GF_BF16);
    {
      int tot = MTOK * DM;
      split_qkv<<<dim3((tot + 255) / 256), dim3(256), 0, stream>>>(big1, Qb, Kb, Vb);
    }
    flash_attn<<<dim3(BATCH * NH, SEQ / 64), dim3(128), 0, stream>>>(Qb, Kb, Vb, att);
    gemm(att, Wolt[l], bol[l], tmpf, nullptr, MTOK, 512, 512, GF_F32);
    ln_kernel<<<dim3(MTOK / 8), dim3(256), 0, stream>>>(tmpf, xf, g1[l], bb1[l], xf, xh, MTOK);
    gemm(xh, Wf1t[l], bf1[l], nullptr, big1, MTOK, 2048, 512, GF_GELU | GF_BF16);
    gemm(big1, Wf2t[l], bf2[l], tmpf, nullptr, MTOK, 512, 2048, GF_F32);
    ln_kernel<<<dim3(MTOK / 8), dim3(256), 0, stream>>>(tmpf, xf, g2[l], bb2[l], xf, xh, MTOK);
  }

  // ---- prediction head + loss ----
  {
    int tot = MZ * DM;
    gather_z<<<dim3((tot + 255) / 256), dim3(256), 0, stream>>>(xh, zh);
  }
  gemm(zh, Wp0t, pb0, nullptr, p1, MZ, 2048, 512,  GF_GELU | GF_BF16);
  gemm(p1, Wp1t, pb1, nullptr, p2, MZ, 2048, 2048, GF_GELU | GF_BF16);
  pred_loss<<<dim3(MZ), dim3(256), 0, stream>>>(p2, pWo, pbo, yt, lp);
  reduce_loss<<<dim3(1), dim3(256), 0, stream>>>(lp, (float*)d_out);
}
